// Jamba4Rec_30399778521708
// MI455X (gfx1250) — matrix operand order repaired and checked by reading, not compiled, not run on hardware
//
#include <hip/hip_runtime.h>

typedef __attribute__((ext_vector_type(16))) _Float16 v16h;
typedef __attribute__((ext_vector_type(8)))  _Float16 v8h;
typedef __attribute__((ext_vector_type(8)))  float    v8f;

#define BATCH   128
#define SEQ     200
#define DMODEL  128
#define DINNER  256
#define DSTATE  32
#define DTRANK  8
#define NHEADS  4
#define HDIM    32
#define ROWS    (BATCH * SEQ)

// ---------------------------------------------------------------------------
// f32 -> f16 bulk conversion (weights; once per launch, ~1MB -> L2-resident)
// ---------------------------------------------------------------------------
__global__ __launch_bounds__(256)
void cvt_f16_kernel(const float* __restrict__ src, _Float16* __restrict__ dst, int n) {
  int i = blockIdx.x * 256 + threadIdx.x;
  if (i < n) dst[i] = (_Float16)src[i];
}

// ---------------------------------------------------------------------------
// Embedding gather + pre-norm LayerNorm.  One wave per (b,l) row. f32 out.
// ---------------------------------------------------------------------------
__global__ __launch_bounds__(256)
void embed_ln_kernel(const int* __restrict__ seqs, const float* __restrict__ emb,
                     const float* __restrict__ pe, const float* __restrict__ w,
                     const float* __restrict__ b, float* __restrict__ out) {
  int wave = threadIdx.x >> 5, lane = threadIdx.x & 31;
  int row = blockIdx.x * 8 + wave;
  if (row >= ROWS) return;
  int l = row % SEQ;
  int item = seqs[row];
  float v[4]; float s = 0.f;
#pragma unroll
  for (int i = 0; i < 4; ++i) {
    int c = lane + 32 * i;
    v[i] = emb[(size_t)item * DMODEL + c] * 11.313708498984761f + pe[l * DMODEL + c];
    s += v[i];
  }
#pragma unroll
  for (int off = 16; off >= 1; off >>= 1) s += __shfl_xor(s, off, 32);
  float mu = s * (1.f / DMODEL);
  float vs = 0.f;
#pragma unroll
  for (int i = 0; i < 4; ++i) { float d = v[i] - mu; vs += d * d; }
#pragma unroll
  for (int off = 16; off >= 1; off >>= 1) vs += __shfl_xor(vs, off, 32);
  float inv = rsqrtf(vs * (1.f / DMODEL) + 1e-5f);
#pragma unroll
  for (int i = 0; i < 4; ++i) {
    int c = lane + 32 * i;
    out[(size_t)row * DMODEL + c] = (v[i] - mu) * inv * w[c] + b[c];
  }
}

// ---------------------------------------------------------------------------
// LayerNorm over D=128, templated output precision (f16 when feeding WMMA).
// ---------------------------------------------------------------------------
template <typename TO>
__global__ __launch_bounds__(256)
void ln_kernel(const float* __restrict__ in, const float* __restrict__ w,
               const float* __restrict__ b, TO* __restrict__ out, float eps) {
  int wave = threadIdx.x >> 5, lane = threadIdx.x & 31;
  int row = blockIdx.x * 8 + wave;
  if (row >= ROWS) return;
  const float* xr = in + (size_t)row * DMODEL;
  float v[4]; float s = 0.f;
#pragma unroll
  for (int i = 0; i < 4; ++i) { v[i] = xr[lane + 32 * i]; s += v[i]; }
#pragma unroll
  for (int off = 16; off >= 1; off >>= 1) s += __shfl_xor(s, off, 32);
  float mu = s * (1.f / DMODEL);
  float vs = 0.f;
#pragma unroll
  for (int i = 0; i < 4; ++i) { float d = v[i] - mu; vs += d * d; }
#pragma unroll
  for (int off = 16; off >= 1; off >>= 1) vs += __shfl_xor(vs, off, 32);
  float inv = rsqrtf(vs * (1.f / DMODEL) + eps);
#pragma unroll
  for (int i = 0; i < 4; ++i) {
    int c = lane + 32 * i;
    out[(size_t)row * DMODEL + c] = (TO)((v[i] - mu) * inv * w[c] + b[c]);
  }
}

// ---------------------------------------------------------------------------
// A-fragment loaders: f16 path = two 16B vector loads; f32 path converts.
// Layout (16x32 f16 A): lane ln -> row; lane-half hi selects K quarters.
// ---------------------------------------------------------------------------
__device__ __forceinline__ v16h load_afrag(const _Float16* A, size_t off) {
  const v8h* p = (const v8h*)(A + off);
  v8h lo = p[0], hh = p[2];
  v16h r;
#pragma unroll
  for (int i = 0; i < 8; ++i) { r[i] = lo[i]; r[i + 8] = hh[i]; }
  return r;
}
__device__ __forceinline__ v16h load_afrag(const float* A, size_t off) {
  const float* p = A + off;
  v16h r;
#pragma unroll
  for (int i = 0; i < 8; ++i) { r[i] = (_Float16)p[i]; r[i + 8] = (_Float16)p[i + 16]; }
  return r;
}

// ---------------------------------------------------------------------------
// WMMA GEMM: out[M,N] = epi(A[M,K] @ W[N,K]^T), W pre-converted to f16.
//   EPI 0: none    1: +res    2: +bias, relu    3: +bias, +res
// Block = 4 waves; wave computes 16(M) x 64(N); grid=(ceil(N/64), M/64).
// ---------------------------------------------------------------------------
template <int EPI, typename TA, typename TO>
__global__ __launch_bounds__(128)
void gemm_wmma_kernel(const TA* __restrict__ A, const _Float16* __restrict__ W,
                      const float* __restrict__ bias, const float* __restrict__ res,
                      TO* __restrict__ out, int M, int N, int K) {
  int lane = threadIdx.x & 31, wave = threadIdx.x >> 5;
  int hi = lane >> 4, ln = lane & 15;
  int m0 = blockIdx.y * 64 + wave * 16;
  int n0 = blockIdx.x * 64;
  v8f zero = {0.f, 0.f, 0.f, 0.f, 0.f, 0.f, 0.f, 0.f};
  v8f acc[4] = {zero, zero, zero, zero};
  int ma = m0 + ln;

  for (int k0 = 0; k0 < K; k0 += 32) {
    size_t aoff = (size_t)ma * K + k0 + hi * 8;
    if (k0 + 32 < K) __builtin_prefetch(A + aoff + 32, 0, 1);
    v16h af = load_afrag(A, aoff);
#pragma unroll
    for (int t = 0; t < 4; ++t) {
      int n = n0 + t * 16 + ln;
      v16h bf;
      if (n < N) {
        const v8h* wp = (const v8h*)(W + (size_t)n * K + k0 + hi * 8);
        v8h b0 = wp[0], b1 = wp[2];
#pragma unroll
        for (int i = 0; i < 8; ++i) { bf[i] = b0[i]; bf[i + 8] = b1[i]; }
      } else {
#pragma unroll
        for (int i = 0; i < 16; ++i) bf[i] = (_Float16)0.f;
      }
      acc[t] = __builtin_amdgcn_wmma_f32_16x16x32_f16(false, af, false, bf,
                                                      (short)0, acc[t], false, false);
    }
  }

#pragma unroll
  for (int t = 0; t < 4; ++t) {
    int col = n0 + t * 16 + ln;
    if (col >= N) continue;
#pragma unroll
    for (int r = 0; r < 8; ++r) {
      int row = m0 + r + 8 * hi;
      float v = acc[t][r];
      if (EPI == 2) { v += bias[col]; v = v > 0.f ? v : 0.f; }
      if (EPI == 3) { v += bias[col] + res[(size_t)row * N + col]; }
      if (EPI == 1) { v += res[(size_t)row * N + col]; }
      out[(size_t)row * N + col] = (TO)v;
    }
  }
}

// ---------------------------------------------------------------------------
// Depthwise causal conv (width 4) + SiLU; writes f32 (for scan) + f16 (GEMM).
// ---------------------------------------------------------------------------
__global__ __launch_bounds__(256)
void conv_silu_kernel(const float* __restrict__ xz, const float* __restrict__ cw,
                      const float* __restrict__ cb, float* __restrict__ xc,
                      _Float16* __restrict__ xch) {
  int idx = blockIdx.x * 256 + threadIdx.x;
  int d = idx & (DINNER - 1);
  int row = idx >> 8;
  int l = row % SEQ;
  float acc = cb[d];
#pragma unroll
  for (int kk = 0; kk < 4; ++kk) {
    int ls = l + kk - 3;
    if (ls >= 0) acc += cw[d * 4 + kk] * xz[(size_t)(row + kk - 3) * (2 * DINNER) + d];
  }
  float sv = acc * (1.f / (1.f + __expf(-acc)));
  xc[(size_t)row * DINNER + d] = sv;
  xch[(size_t)row * DINNER + d] = (_Float16)sv;
}

// ---------------------------------------------------------------------------
// dt = softplus(dbl[:, :8] @ dt_w^T + dt_b), per (row, d).
// ---------------------------------------------------------------------------
__global__ __launch_bounds__(256)
void dt_kernel(const float* __restrict__ dbl, const float* __restrict__ dtw,
               const float* __restrict__ dtb, float* __restrict__ dt) {
  int idx = blockIdx.x * 256 + threadIdx.x;
  int d = idx & (DINNER - 1);
  int row = idx >> 8;
  float a = dtb[d];
#pragma unroll
  for (int r = 0; r < DTRANK; ++r) a += dbl[(size_t)row * 72 + r] * dtw[d * DTRANK + r];
  float sp = (a > 20.f) ? a : log1pf(__expf(a));
  dt[(size_t)row * DINNER + d] = sp;
}

// ---------------------------------------------------------------------------
// Selective scan.  Block per batch element; thread = channel d; 32-state in
// registers.  B_t/C_t broadcast into LDS with GLOBAL_LOAD_ASYNC_TO_LDS_B32
// (ASYNCcnt-tracked), waited with s_wait_asynccnt before the barrier.
// Fuses y = (scan + D*x) * silu(z); writes f16 for the out-proj WMMA GEMM.
// ---------------------------------------------------------------------------
__global__ __launch_bounds__(256)
void scan_kernel(const float* __restrict__ dbl, const float* __restrict__ dt,
                 const float* __restrict__ xc, const float* __restrict__ xz,
                 const float* __restrict__ Alog, const float* __restrict__ Dp,
                 _Float16* __restrict__ y) {
  __shared__ float BC[64];
  int b = blockIdx.x;
  int d = threadIdx.x;
  float h[DSTATE];
  float Ar[DSTATE];
#pragma unroll
  for (int s = 0; s < DSTATE; ++s) {
    h[s] = 0.f;
    Ar[s] = -__expf(Alog[(size_t)d * DSTATE + s]);
  }
  float Dd = Dp[d];
  for (int l = 0; l < SEQ; ++l) {
    size_t row = (size_t)b * SEQ + l;
    if (threadIdx.x < 64) {
      // async copy of this step's B (32) and C (32) vectors into LDS
      unsigned long long ga =
          (unsigned long long)(const void*)(dbl + row * 72 + 8 + threadIdx.x);
      unsigned int lo = (unsigned int)(unsigned long long)(const void*)(&BC[threadIdx.x]);
      asm volatile("global_load_async_to_lds_b32 %0, %1, off"
                   :: "v"(lo), "v"(ga) : "memory");
      asm volatile("s_wait_asynccnt 0x0" ::: "memory");
    }
    __syncthreads();
    float dtv = dt[row * DINNER + d];
    float xv = xc[row * DINNER + d];
    float dx = dtv * xv;
    float yv = 0.f;
#pragma unroll
    for (int s = 0; s < DSTATE; ++s) {
      float dA = __expf(dtv * Ar[s]);
      h[s] = dA * h[s] + dx * BC[s];
      yv += h[s] * BC[32 + s];
    }
    float zv = xz[row * (2 * DINNER) + DINNER + d];
    float g = zv * (1.f / (1.f + __expf(-zv)));
    y[row * DINNER + d] = (_Float16)((yv + Dd * xv) * g);
    __syncthreads();
  }
}

// ---------------------------------------------------------------------------
// Causal attention core via WMMA on f16 q/k/v.  One wave per (q-tile, b, h).
// ---------------------------------------------------------------------------
#define KT 13
__global__ __launch_bounds__(32)
void attn_kernel(const _Float16* __restrict__ q, const _Float16* __restrict__ k,
                 const _Float16* __restrict__ v, _Float16* __restrict__ o) {
  __shared__ float    S[16 * 208];
  __shared__ _Float16 P[16 * 224];
  int mt = blockIdx.x;
  int bh = blockIdx.y;
  int b = bh >> 2, h = bh & 3;
  int lane = threadIdx.x;
  int hi = lane >> 4, ln = lane & 15;
  size_t base = (size_t)b * SEQ * DMODEL + h * HDIM;
  v8f zero = {0.f, 0.f, 0.f, 0.f, 0.f, 0.f, 0.f, 0.f};

  int qr = mt * 16 + ln;
  v16h af;
  if (qr < SEQ) {
    af = load_afrag(q, base + (size_t)qr * DMODEL + hi * 8);
  } else {
#pragma unroll
    for (int i = 0; i < 16; ++i) af[i] = (_Float16)0.f;
  }

  const float scale = 0.17677669529663689f;  // 32^-0.5
  for (int j = 0; j < KT; ++j) {
    int kidx = j * 16 + ln;
    v16h bf;
    if (kidx < SEQ) {
      const v8h* kp = (const v8h*)(k + base + (size_t)kidx * DMODEL + hi * 8);
      v8h b0 = kp[0], b1 = kp[2];
#pragma unroll
      for (int i = 0; i < 8; ++i) { bf[i] = b0[i]; bf[i + 8] = b1[i]; }
    } else {
#pragma unroll
      for (int i = 0; i < 16; ++i) bf[i] = (_Float16)0.f;
    }
    v8f s = __builtin_amdgcn_wmma_f32_16x16x32_f16(false, af, false, bf,
                                                   (short)0, zero, false, false);
#pragma unroll
    for (int r = 0; r < 8; ++r) {
      int m = r + 8 * hi;
      int qi = mt * 16 + m;
      int ki = j * 16 + ln;
      float val = (ki <= qi && ki < SEQ) ? s[r] * scale : -1e9f;
      S[m * 208 + ki] = val;
    }
  }
  __syncthreads();

  if (lane < 16) {
    int m = lane;
    int qi = mt * 16 + m;
    if (qi < SEQ) {
      float mx = -3.4e38f;
      for (int c = 0; c < 208; ++c) { float t = S[m * 208 + c]; mx = t > mx ? t : mx; }
      float sum = 0.f;
      for (int c = 0; c < 208; ++c) {
        float e = __expf(S[m * 208 + c] - mx);
        S[m * 208 + c] = e;
        sum += e;
      }
      float inv = 1.f / sum;
      for (int c = 0; c < 208; ++c) P[m * 224 + c] = (_Float16)(S[m * 208 + c] * inv);
      for (int c = 208; c < 224; ++c) P[m * 224 + c] = (_Float16)0.f;
    } else {
      for (int c = 0; c < 224; ++c) P[m * 224 + c] = (_Float16)0.f;
    }
  }
  __syncthreads();

  v8f acc[2] = {zero, zero};
  for (int kc = 0; kc < 7; ++kc) {
    int kb = kc * 32;
    v16h pa;
    const _Float16* pr = &P[ln * 224 + kb + hi * 16];
#pragma unroll
    for (int i = 0; i < 8; ++i) { pa[i] = pr[i]; pa[i + 8] = pr[i + 8]; }
#pragma unroll
    for (int t = 0; t < 2; ++t) {
      int n = t * 16 + ln;
      int koff = kb + hi * 16;
      v16h vf;
#pragma unroll
      for (int i = 0; i < 16; ++i) {
        int key = koff + i;
        vf[i] = (key < SEQ) ? v[base + (size_t)key * DMODEL + n] : (_Float16)0.f;
      }
      acc[t] = __builtin_amdgcn_wmma_f32_16x16x32_f16(false, pa, false, vf,
                                                      (short)0, acc[t], false, false);
    }
  }
#pragma unroll
  for (int t = 0; t < 2; ++t)
#pragma unroll
    for (int r = 0; r < 8; ++r) {
      int m = r + 8 * hi;
      int qi = mt * 16 + m;
      if (qi < SEQ) o[base + (size_t)qi * DMODEL + t * 16 + ln] = (_Float16)acc[t][r];
    }
}

// ---------------------------------------------------------------------------
// Final logits: pos/neg dot products.  One wave per row.
// ---------------------------------------------------------------------------
__global__ __launch_bounds__(256)
void logits_kernel(const float* __restrict__ x, const float* __restrict__ emb,
                   const int* __restrict__ pos, const int* __restrict__ neg,
                   float* __restrict__ out) {
  int wave = threadIdx.x >> 5, lane = threadIdx.x & 31;
  int row = blockIdx.x * 8 + wave;
  if (row >= ROWS) return;
  int pi = pos[row], ni = neg[row];
  float sp = 0.f, sn = 0.f;
#pragma unroll
  for (int i = 0; i < 4; ++i) {
    int c = lane + 32 * i;
    float xv = x[(size_t)row * DMODEL + c];
    sp += xv * emb[(size_t)pi * DMODEL + c];
    sn += xv * emb[(size_t)ni * DMODEL + c];
  }
#pragma unroll
  for (int off = 16; off >= 1; off >>= 1) {
    sp += __shfl_xor(sp, off, 32);
    sn += __shfl_xor(sn, off, 32);
  }
  if (lane == 0) {
    out[row] = sp;
    out[ROWS + row] = sn;
  }
}

// ---------------------------------------------------------------------------
// Host orchestration
// ---------------------------------------------------------------------------
static inline dim3 gemm_grid(int M, int N) { return dim3((N + 63) / 64, (M + 63) / 64); }

extern "C" void kernel_launch(void* const* d_in, const int* in_sizes, int n_in,
                              void* d_out, int out_size, void* d_ws, size_t ws_size,
                              hipStream_t stream) {
  const int*   log_seqs = (const int*)d_in[1];
  const int*   pos_seqs = (const int*)d_in[2];
  const int*   neg_seqs = (const int*)d_in[3];
  const float* item_emb = (const float*)d_in[4];
  const float* pos_emb  = (const float*)d_in[5];
  const float* pre_w    = (const float*)d_in[6];
  const float* pre_b    = (const float*)d_in[7];
  const float* m_in_w   = (const float*)d_in[8];
  const float* m_conv_w = (const float*)d_in[9];
  const float* m_conv_b = (const float*)d_in[10];
  const float* m_xp_w   = (const float*)d_in[11];
  const float* m_dt_w   = (const float*)d_in[12];
  const float* m_dt_b   = (const float*)d_in[13];
  const float* m_Alog   = (const float*)d_in[14];
  const float* m_D      = (const float*)d_in[15];
  const float* m_out_w  = (const float*)d_in[16];
  const float* ln1_w    = (const float*)d_in[17];
  const float* ln1_b    = (const float*)d_in[18];
  const float* ln2_w    = (const float*)d_in[19];
  const float* ln2_b    = (const float*)d_in[20];
  const float* wq       = (const float*)d_in[21];
  const float* wk       = (const float*)d_in[22];
  const float* wv       = (const float*)d_in[23];
  const float* wo       = (const float*)d_in[24];
  const float* ffn_w1   = (const float*)d_in[25];
  const float* ffn_b1   = (const float*)d_in[26];
  const float* ffn_w2   = (const float*)d_in[27];
  const float* ffn_b2   = (const float*)d_in[28];
  const float* ffln_w   = (const float*)d_in[29];
  const float* ffln_b   = (const float*)d_in[30];
  float* out = (float*)d_out;

  const size_t R = ROWS;
  float* wsf = (float*)d_ws;
  size_t o = 0;
  float* xbuf = wsf + o; o += R * 128;   // residual stream (f32)
  float* xzb  = wsf + o; o += R * 512;   // xz (f32: conv + z gate)
  float* xcb  = wsf + o; o += R * 256;   // conv output (f32 for scan)
  float* dtb  = wsf + o; o += R * 256;   // dt (f32)
  float* dblb = wsf + o; o += R * 72;    // x-proj output (f32)
  float* tmpf = wsf + o; o += R * 128;   // ff + x (f32)
  float* xfin = wsf + o; o += R * 128;   // final LN output (f32)

  _Float16* hb = (_Float16*)(wsf + o);
  size_t ho = 0;
  _Float16* tmp16 = hb + ho; ho += R * 128;   // LN output (GEMM A)
  _Float16* xc16  = hb + ho; ho += R * 256;   // conv output (GEMM A)
  _Float16* sh16  = hb + ho; ho += R * 512;   // y16 / q|k|v|o16 / ff1
  _Float16* w16   = hb + ho;
  _Float16* y16 = sh16;
  _Float16* q16 = sh16;
  _Float16* k16 = sh16 + R * 128;
  _Float16* v16b = sh16 + R * 256;
  _Float16* o16 = sh16 + R * 384;
  _Float16* ff16 = sh16;

  // f16 weight pool
  _Float16* w_in = w16;                 // 2*512*128
  _Float16* w_xp = w_in + 131072;       // 2*72*256
  _Float16* w_ou = w_xp + 36864;        // 2*128*256
  _Float16* w_q  = w_ou + 65536;        // 2*128*128
  _Float16* w_k  = w_q + 32768;
  _Float16* w_v  = w_k + 32768;
  _Float16* w_o  = w_v + 32768;
  _Float16* w_f1 = w_o + 32768;         // 512*128
  _Float16* w_f2 = w_f1 + 65536;        // 128*512

  dim3 blk256(256), blk128(128), blk32(32);
  int rows8 = ROWS / 8;

  // ---- one-time (per launch) weight conversion ----
  cvt_f16_kernel<<<(131072 + 255) / 256, blk256, 0, stream>>>(m_in_w, w_in, 131072);
  cvt_f16_kernel<<<(36864 + 255) / 256, blk256, 0, stream>>>(m_xp_w, w_xp, 36864);
  cvt_f16_kernel<<<(65536 + 255) / 256, blk256, 0, stream>>>(m_out_w, w_ou, 65536);
  cvt_f16_kernel<<<(32768 + 255) / 256, blk256, 0, stream>>>(wq, w_q, 32768);
  cvt_f16_kernel<<<(32768 + 255) / 256, blk256, 0, stream>>>(wk, w_k, 32768);
  cvt_f16_kernel<<<(32768 + 255) / 256, blk256, 0, stream>>>(wv, w_v, 32768);
  cvt_f16_kernel<<<(32768 + 255) / 256, blk256, 0, stream>>>(wo, w_o, 32768);
  cvt_f16_kernel<<<(65536 + 255) / 256, blk256, 0, stream>>>(ffn_w1, w_f1, 65536);
  cvt_f16_kernel<<<(65536 + 255) / 256, blk256, 0, stream>>>(ffn_w2, w_f2, 65536);

  embed_ln_kernel<<<rows8, blk256, 0, stream>>>(log_seqs, item_emb, pos_emb, pre_w, pre_b, xbuf);

  for (int l = 0; l < 2; ++l) {
    const float* cw   = m_conv_w + (size_t)l * 256 * 4;
    const float* cb   = m_conv_b + (size_t)l * 256;
    const float* dtw  = m_dt_w + (size_t)l * 256 * 8;
    const float* dtbi = m_dt_b + (size_t)l * 256;
    const float* Al   = m_Alog + (size_t)l * 256 * 32;
    const float* Dl   = m_D + (size_t)l * 256;

    // ---- Mamba block ----
    ln_kernel<_Float16><<<rows8, blk256, 0, stream>>>(xbuf, ln1_w + l * 128, ln1_b + l * 128, tmp16, 1e-5f);
    gemm_wmma_kernel<0, _Float16, float><<<gemm_grid(ROWS, 512), blk128, 0, stream>>>(
        tmp16, w_in + (size_t)l * 512 * 128, nullptr, nullptr, xzb, ROWS, 512, 128);
    conv_silu_kernel<<<ROWS * DINNER / 256, blk256, 0, stream>>>(xzb, cw, cb, xcb, xc16);
    gemm_wmma_kernel<0, _Float16, float><<<gemm_grid(ROWS, 72), blk128, 0, stream>>>(
        xc16, w_xp + (size_t)l * 72 * 256, nullptr, nullptr, dblb, ROWS, 72, 256);
    dt_kernel<<<ROWS * DINNER / 256, blk256, 0, stream>>>(dblb, dtw, dtbi, dtb);
    scan_kernel<<<BATCH, blk256, 0, stream>>>(dblb, dtb, xcb, xzb, Al, Dl, y16);
    gemm_wmma_kernel<1, _Float16, float><<<gemm_grid(ROWS, 128), blk128, 0, stream>>>(
        y16, w_ou + (size_t)l * 128 * 256, nullptr, xbuf, xbuf, ROWS, 128, 256);

    // ---- Attention block ----
    ln_kernel<_Float16><<<rows8, blk256, 0, stream>>>(xbuf, ln2_w + l * 128, ln2_b + l * 128, tmp16, 1e-5f);
    gemm_wmma_kernel<0, _Float16, _Float16><<<gemm_grid(ROWS, 128), blk128, 0, stream>>>(
        tmp16, w_q + (size_t)l * 128 * 128, nullptr, nullptr, q16, ROWS, 128, 128);
    gemm_wmma_kernel<0, _Float16, _Float16><<<gemm_grid(ROWS, 128), blk128, 0, stream>>>(
        tmp16, w_k + (size_t)l * 128 * 128, nullptr, nullptr, k16, ROWS, 128, 128);
    gemm_wmma_kernel<0, _Float16, _Float16><<<gemm_grid(ROWS, 128), blk128, 0, stream>>>(
        tmp16, w_v + (size_t)l * 128 * 128, nullptr, nullptr, v16b, ROWS, 128, 128);
    attn_kernel<<<dim3(KT, BATCH * NHEADS), blk32, 0, stream>>>(q16, k16, v16b, o16);
    gemm_wmma_kernel<1, _Float16, float><<<gemm_grid(ROWS, 128), blk128, 0, stream>>>(
        o16, w_o + (size_t)l * 128 * 128, nullptr, xbuf, xbuf, ROWS, 128, 128);
  }

  // ---- FFN + final LN ----
  gemm_wmma_kernel<2, float, _Float16><<<gemm_grid(ROWS, 512), blk128, 0, stream>>>(
      xbuf, w_f1, ffn_b1, nullptr, ff16, ROWS, 512, 128);
  gemm_wmma_kernel<3, _Float16, float><<<gemm_grid(ROWS, 128), blk128, 0, stream>>>(
      ff16, w_f2, ffn_b2, xbuf, tmpf, ROWS, 128, 512);
  ln_kernel<float><<<rows8, blk256, 0, stream>>>(tmpf, ffln_w, ffln_b, xfin, 1e-6f);

  logits_kernel<<<rows8, blk256, 0, stream>>>(xfin, item_emb, pos_seqs, neg_seqs, out);
}
